// Linear_Embedding_Dimensional_Attention_49701361549933
// MI455X (gfx1250) — matrix operand order repaired and checked by reading, not compiled, not run on hardware
//
#include <hip/hip_runtime.h>
#include <hip/hip_bf16.h>
#include <math.h>

// Problem constants
#define BB   16
#define NN   4096
#define CC   384
#define HH   8
#define NHH  512
#define MROWS (BB*CC)   // 6144
#define K3   (3*NN)     // 12288
#define KSTEPS (NN/32)  // 128

typedef __attribute__((ext_vector_type(16))) __bf16 bf16x16;
typedef __attribute__((ext_vector_type(8)))  float  v8f;

struct alignas(16) B128 { unsigned int u[4]; };
union Frag { bf16x16 v; B128 b[2]; };

// LDS layout per buffer: A tile 128 rows x 40 bf16 (padded) = 10240 B,
//                        B tile 256 rows x 40 bf16 (padded) = 20480 B
#define ABUF_BYTES 10240u
#define BUF_BYTES  30720u
#define SMEM_BYTES 61440u   // double buffered

__device__ __forceinline__ unsigned short f2bf(float f) {
    unsigned int u = __float_as_uint(f);
    unsigned int r = u + 0x7FFFu + ((u >> 16) & 1u);
    return (unsigned short)(r >> 16);
}

__device__ __forceinline__ void async_b128(unsigned lds_off, unsigned long long gaddr) {
    // CDNA5 async global->LDS copy, tracked by ASYNCcnt (CDNA5 ISA)
    asm volatile("global_load_async_to_lds_b128 %0, %1, off"
                 :: "v"(lds_off), "v"(gaddr) : "memory");
}

// ---------------------------------------------------------------------------
// Shared WMMA mainloop: acc[4][4] (64x64 wave tile), block tile 128M x 256N,
// K-step 32, async double-buffered LDS staging.
// ---------------------------------------------------------------------------
__device__ __forceinline__ void wmma_mainloop(
    const unsigned short* __restrict__ A,   // bf16 [*, NN] row-major
    const unsigned short* __restrict__ W,   // bf16 [*, NN] row-major
    int m0, int n0, char* smem, v8f (&acc)[4][4])
{
    const int t    = threadIdx.x;
    const int lane = t & 31, w = t >> 5;
    const int wr   = w >> 2, wc = w & 3;        // 2 x 4 waves
    const int r    = lane & 15;
    const int hi   = lane >> 4;
    const int rA   = t >> 2;                    // staging row 0..63
    const int off16= t & 3;                     // 16B chunk in 64B row segment

    // global byte addresses for this thread's 6 chunks (advance 64B per K-step)
    unsigned long long ga0 = (unsigned long long)(const void*)(A + (size_t)(m0 + rA) * NN + off16 * 8);
    unsigned long long ga1 = ga0 + (unsigned long long)64 * NN * 2;
    unsigned long long gb0 = (unsigned long long)(const void*)(W + (size_t)(n0 + rA) * NN + off16 * 8);
    unsigned long long gb1 = gb0 + (unsigned long long)64  * NN * 2;
    unsigned long long gb2 = gb0 + (unsigned long long)128 * NN * 2;
    unsigned long long gb3 = gb0 + (unsigned long long)192 * NN * 2;

    // LDS byte offsets (low 32 bits of generic pointer = LDS address)
    const unsigned sb = (unsigned)(size_t)(void*)smem;
    const unsigned la = sb + (unsigned)(rA * 40 + off16 * 8) * 2;
    const unsigned lb = sb + ABUF_BYTES + (unsigned)(rA * 40 + off16 * 8) * 2;

    auto stage = [&](int buf) {
        const unsigned o = (unsigned)buf * BUF_BYTES;
        async_b128(la + o,          ga0);
        async_b128(la + o + 5120u,  ga1);
        async_b128(lb + o,          gb0);
        async_b128(lb + o + 5120u,  gb1);
        async_b128(lb + o + 10240u, gb2);
        async_b128(lb + o + 15360u, gb3);
        ga0 += 64; ga1 += 64; gb0 += 64; gb1 += 64; gb2 += 64; gb3 += 64;
    };

    stage(0);
    #pragma unroll 2
    for (int it = 0; it < KSTEPS; ++it) {
        const int cur = it & 1;
        if (it + 1 < KSTEPS) {
            stage(cur ^ 1);
            asm volatile("s_wait_asynccnt 6" ::: "memory");   // current buffer's 6 done
        } else {
            asm volatile("s_wait_asynccnt 0" ::: "memory");
        }
        __syncthreads();

        const unsigned short* Asp = (const unsigned short*)(smem + cur * BUF_BYTES);
        const unsigned short* Bsp = Asp + ABUF_BYTES / 2;     // bf16 elements

        Frag af[4];
        #pragma unroll
        for (int mi = 0; mi < 4; mi++) {      // A frag: K chunks {hi*8..}, {hi*8+16..}
            const unsigned short* p = Asp + (wr * 64 + mi * 16 + r) * 40 + hi * 8;
            af[mi].b[0] = *(const B128*)p;
            af[mi].b[1] = *(const B128*)(p + 16);
        }
        #pragma unroll
        for (int ni = 0; ni < 4; ni++) {      // B frag: K chunks {hi*8..}, {hi*8+16..}
            Frag bg;
            const unsigned short* p = Bsp + (wc * 64 + ni * 16 + r) * 40 + hi * 8;
            bg.b[0] = *(const B128*)p;
            bg.b[1] = *(const B128*)(p + 16);
            #pragma unroll
            for (int mi = 0; mi < 4; mi++)
                acc[mi][ni] = __builtin_amdgcn_wmma_f32_16x16x32_bf16(
                    false, af[mi].v, false, bg.v, (short)0, acc[mi][ni], false, false);
        }
        __syncthreads();
    }
}

// ---------------------------------------------------------------------------
// x [B,N,C] f32  ->  xt [B*C, N] bf16   (transpose via LDS tile)
// ---------------------------------------------------------------------------
__global__ __launch_bounds__(256) void xt_cast_kernel(
    const float* __restrict__ x, unsigned short* __restrict__ xt)
{
    __shared__ float tile[32][33];
    const int b = blockIdx.z;
    const int n0 = blockIdx.x * 32, c0 = blockIdx.y * 32;
    const int tx = threadIdx.x, ty = threadIdx.y;   // (32, 8)
    #pragma unroll
    for (int j = 0; j < 4; j++) {
        int n = n0 + ty + j * 8;
        tile[ty + j * 8][tx] = x[(size_t)b * NN * CC + (size_t)n * CC + (c0 + tx)];
    }
    __syncthreads();
    #pragma unroll
    for (int j = 0; j < 4; j++) {
        int c = c0 + ty + j * 8;
        xt[(size_t)(b * CC + c) * NN + (n0 + tx)] = f2bf(tile[tx][ty + j * 8]);
    }
}

// ---------------------------------------------------------------------------
// generic f32 -> bf16 cast
// ---------------------------------------------------------------------------
__global__ __launch_bounds__(256) void cast_bf16_kernel(
    const float* __restrict__ src, unsigned short* __restrict__ dst, int n)
{
    int i = blockIdx.x * 256 + threadIdx.x;
    if (i < n) dst[i] = f2bf(src[i]);
}

// ---------------------------------------------------------------------------
// WMMA GEMM #1:  Out[6144, 12288] = A[6144, 4096] * W[12288, 4096]^T
// ---------------------------------------------------------------------------
__global__ __launch_bounds__(256) void gemm_qkv_kernel(
    const unsigned short* __restrict__ A,
    const unsigned short* __restrict__ W,
    float* __restrict__ Out)
{
    __shared__ __align__(16) char smem[SMEM_BYTES];
    const int m0 = blockIdx.y * 128;
    const int n0 = blockIdx.x * 256;

    v8f acc[4][4];
    #pragma unroll
    for (int i = 0; i < 4; i++)
        #pragma unroll
        for (int j = 0; j < 4; j++)
            #pragma unroll
            for (int e = 0; e < 8; e++) acc[i][j][e] = 0.0f;

    wmma_mainloop(A, W, m0, n0, smem, acc);

    const int t = threadIdx.x, lane = t & 31, w = t >> 5;
    const int wr = w >> 2, wc = w & 3;
    const int r = lane & 15, hi = lane >> 4;
    #pragma unroll
    for (int mi = 0; mi < 4; mi++)
        #pragma unroll
        for (int ni = 0; ni < 4; ni++) {
            int gn = n0 + wc * 64 + ni * 16 + r;
            int gm = m0 + wr * 64 + mi * 16 + hi * 8;
            #pragma unroll
            for (int e = 0; e < 8; e++)
                Out[(size_t)(gm + e) * K3 + gn] = acc[mi][ni][e];
        }
}

// ---------------------------------------------------------------------------
// per-(row,h) inverse L2 norms of q and k segments (512 elems each)
// ---------------------------------------------------------------------------
__global__ __launch_bounds__(256) void norm_kernel(
    const float* __restrict__ qkv, float* __restrict__ invq, float* __restrict__ invk)
{
    int task = blockIdx.x * 8 + (threadIdx.x >> 5);
    int lane = threadIdx.x & 31;
    int isK  = task >= MROWS * HH;
    int id   = isK ? task - MROWS * HH : task;
    int row  = id >> 3, h = id & 7;
    const float* p = qkv + (size_t)row * K3 + (isK ? NN : 0) + h * NHH;
    float s = 0.f;
    for (int i = lane; i < NHH; i += 32) { float v = p[i]; s += v * v; }
    #pragma unroll
    for (int m = 16; m >= 1; m >>= 1) s += __shfl_xor(s, m, 32);
    if (lane == 0) (isK ? invk : invq)[row * 8 + h] = rsqrtf(s);
}

// ---------------------------------------------------------------------------
// kernel_trick[b,h,nh] = scale[h] * sum_c  k_hat * v
// ---------------------------------------------------------------------------
__global__ __launch_bounds__(256) void ktrick_kernel(
    const float* __restrict__ qkv, const float* __restrict__ invk,
    const float* __restrict__ scale, float* __restrict__ kt)
{
    int tid = blockIdx.x * 256 + threadIdx.x;   // 0..65535
    int nh = tid & 511, h = (tid >> 9) & 7, b = tid >> 12;
    float s = 0.f;
    for (int c = 0; c < CC; c++) {
        int row = b * CC + c;
        size_t base = (size_t)row * K3;
        float kk = qkv[base + NN     + h * NHH + nh] * invk[row * 8 + h];
        float vv = qkv[base + 2 * NN + h * NHH + nh];
        s += kk * vv;
    }
    kt[tid] = s * scale[h];
}

// ---------------------------------------------------------------------------
// o = bf16( gelu_exact( q_hat * kernel_trick ) )
// ---------------------------------------------------------------------------
__global__ __launch_bounds__(256) void gelu_o_kernel(
    const float* __restrict__ qkv, const float* __restrict__ invq,
    const float* __restrict__ kt, unsigned short* __restrict__ o)
{
    size_t idx = (size_t)blockIdx.x * 256 + threadIdx.x;
    int row = (int)(idx >> 12);
    int n = (int)(idx & 4095);
    int h = n >> 9, nh = n & 511;
    int b = row / CC;
    float q  = qkv[(size_t)row * K3 + n] * invq[row * 8 + h];
    float xv = q * kt[(b * 8 + h) * NHH + nh];
    float g  = 0.5f * xv * (1.f + erff(xv * 0.70710678118654752f));
    o[(size_t)row * NN + n] = f2bf(g);
}

// ---------------------------------------------------------------------------
// WMMA GEMM #2: proj = gelu_o * wproj^T + bproj, transpose-store -> [B,N,C]
// ---------------------------------------------------------------------------
__global__ __launch_bounds__(256) void gemm_proj_kernel(
    const unsigned short* __restrict__ A,    // bf16 [MROWS, NN]
    const unsigned short* __restrict__ W,    // bf16 [NN, NN]
    const float* __restrict__ bias,          // [NN]
    float* __restrict__ OutT)                // f32 [B, N, C]
{
    __shared__ __align__(16) char smem[SMEM_BYTES];
    const int m0 = blockIdx.y * 128;
    const int n0 = blockIdx.x * 256;

    v8f acc[4][4];
    #pragma unroll
    for (int i = 0; i < 4; i++)
        #pragma unroll
        for (int j = 0; j < 4; j++)
            #pragma unroll
            for (int e = 0; e < 8; e++) acc[i][j][e] = 0.0f;

    wmma_mainloop(A, W, m0, n0, smem, acc);

    const int t = threadIdx.x, lane = t & 31, w = t >> 5;
    const int wr = w >> 2, wc = w & 3;
    const int r = lane & 15, hi = lane >> 4;

    // epilogue: bias + transpose via LDS (reuse staging LDS), 4 quarters of 32 c-rows
    float* Cs = (float*)smem;                 // 32 x 257 f32 = 32896 B <= SMEM_BYTES
    const int b = m0 / CC;                    // C=384 divisible by 128 -> single batch
    const int cbase = m0 - b * CC;
    #pragma unroll
    for (int qtr = 0; qtr < 4; qtr++) {
        __syncthreads();
        if (wr == (qtr >> 1)) {
            const int mis = (qtr & 1) * 2;
            #pragma unroll
            for (int mm = 0; mm < 2; mm++) {
                int mi = mis + mm;
                int cl = mm * 16 + hi * 8;    // row within quarter (0..31)
                #pragma unroll
                for (int ni = 0; ni < 4; ni++) {
                    int nl = wc * 64 + ni * 16 + r;
                    #pragma unroll
                    for (int e = 0; e < 8; e++)
                        Cs[(cl + e) * 257 + nl] = acc[mi][ni][e];
                }
            }
        }
        __syncthreads();
        #pragma unroll
        for (int i = 0; i < 32; i++) {
            int idx = t + i * 256;            // 32 x 256 elems
            int cl = idx & 31, ml = idx >> 5;
            int m = n0 + ml, c = cbase + qtr * 32 + cl;
            OutT[(size_t)b * (NN * CC) + (size_t)m * CC + c] = Cs[cl * 257 + ml] + bias[m];
        }
    }
}

// ---------------------------------------------------------------------------
extern "C" void kernel_launch(void* const* d_in, const int* in_sizes, int n_in,
                              void* d_out, int out_size, void* d_ws, size_t ws_size,
                              hipStream_t stream)
{
    const float* x     = (const float*)d_in[0];   // [B, N, C]
    const float* wq    = (const float*)d_in[1];   // [N, N]
    const float* wkv   = (const float*)d_in[2];   // [2N, N]
    const float* wproj = (const float*)d_in[3];   // [N, N]
    const float* bproj = (const float*)d_in[4];   // [N]
    const float* scale = (const float*)d_in[5];   // [H]
    float* out = (float*)d_out;                   // [B, N, C]

    char* ws = (char*)d_ws;
    unsigned short* xt  = (unsigned short*)(ws);                        //  50,331,648 B
    unsigned short* wc  = (unsigned short*)(ws + 50331648ull);          // 100,663,296 B
    unsigned short* wp  = (unsigned short*)(ws + 150994944ull);         //  33,554,432 B
    float*          qkv = (float*)(ws + 184549376ull);                  // 301,989,888 B
    float*          invq= (float*)(ws + 486539264ull);                  //     196,608 B
    float*          invk= (float*)(ws + 486735872ull);                  //     196,608 B
    float*          ktb = (float*)(ws + 486932480ull);                  //     262,144 B
    unsigned short* ob  = (unsigned short*)(ws + 487194624ull);         //  50,331,648 B

    // 1) transpose + cast x -> xt (bf16)
    xt_cast_kernel<<<dim3(NN / 32, CC / 32, BB), dim3(32, 8), 0, stream>>>(x, xt);

    // 2) cast weights to bf16 (wq ‖ wkv concatenated row-wise; wproj separate)
    cast_bf16_kernel<<<(NN * NN + 255) / 256, 256, 0, stream>>>(wq, wc, NN * NN);
    cast_bf16_kernel<<<(2 * NN * NN + 255) / 256, 256, 0, stream>>>(wkv, wc + (size_t)NN * NN, 2 * NN * NN);
    cast_bf16_kernel<<<(NN * NN + 255) / 256, 256, 0, stream>>>(wproj, wp, NN * NN);

    // 3) fused qkv GEMM (WMMA bf16 -> f32), block tile 128x256
    gemm_qkv_kernel<<<dim3(K3 / 256, MROWS / 128), 256, 0, stream>>>(xt, wc, qkv);

    // 4) inverse norms of q and k rows
    norm_kernel<<<(2 * MROWS * HH) / 8, 256, 0, stream>>>(qkv, invq, invk);

    // 5) kernel-trick C-reduction
    ktrick_kernel<<<(BB * HH * NHH) / 256, 256, 0, stream>>>(qkv, invk, scale, ktb);

    // 6) q_hat * kt, exact GELU, cast to bf16
    gelu_o_kernel<<<(size_t)MROWS * NN / 256, 256, 0, stream>>>(qkv, invq, ktb, ob);

    // 7) projection GEMM + bias + transpose store -> [B, N, C]
    gemm_proj_kernel<<<dim3(NN / 256, MROWS / 128), 256, 0, stream>>>(ob, wp, bproj, out);
}
